// MultiHeadAttention_31121333027145
// MI455X (gfx1250) — matrix operand order repaired and checked by reading, not compiled, not run on hardware
//
#include <hip/hip_runtime.h>
#include <hip/hip_bf16.h>
#include <type_traits>

// ---------------------------------------------------------------------------
// MI455X (gfx1250, wave32) MHA forward: bf16 WMMA GEMMs + register-resident
// flash attention (S^T trick: P stays in registers, V pre-transposed,
// 32 queries/wave, masked work confined to the diagonal tile).
// ---------------------------------------------------------------------------

typedef __attribute__((ext_vector_type(4)))  __bf16 v4bf;
typedef __attribute__((ext_vector_type(8)))  __bf16 v8bf;
typedef __attribute__((ext_vector_type(16))) __bf16 v16bf;
typedef __attribute__((ext_vector_type(8)))  float  v8f;

static constexpr int BATCH = 2;
static constexpr int SEQ   = 2048;
static constexpr int DM    = 1024;
static constexpr int NH    = 16;
static constexpr int DK    = 64;

#define DEVI __device__ __forceinline__

DEVI v8f wmma_bf16(v16bf a, v16bf b, v8f c) {
  // D = A(16x32 bf16) * B(32x16 bf16) + C(16x16 f32)
  return __builtin_amdgcn_wmma_f32_16x16x32_bf16(
      /*neg_a=*/false, a, /*neg_b=*/false, b,
      /*c_mod=*/(short)0, c, /*reuse_a=*/false, /*reuse_b=*/false);
}

DEVI v16bf cat16(v8bf lo, v8bf hi) {
  return __builtin_shufflevector(lo, hi, 0, 1, 2, 3, 4, 5, 6, 7,
                                 8, 9, 10, 11, 12, 13, 14, 15);
}

// A-fragment (16x32, row-major source: rows = M, cols = K).
// Lane l: m = l&15; halves [0..7] = K[kb..kb+7], [8..15] = K[16+kb..16+kb+7],
// kb = (l<16) ? 0 : 8.   (ISA 7.12.2, 16-bit A 16x32 table)
DEVI v16bf load_fragA(const __bf16* p, int ld) {
  int lane = threadIdx.x & 31;
  const __bf16* row = p + (size_t)(lane & 15) * ld + ((lane >> 4) ? 8 : 0);
  return cat16(*(const v8bf*)(row), *(const v8bf*)(row + 16));
}

// B-fragment (32x16, source indexed rows = N, cols = K, row-major).
// Lane l: n = l&15; halves [0..7] = K[kb..kb+7], [8..15] = K[16+kb..16+kb+7],
// kb = (l<16) ? 0 : 8.
DEVI v16bf load_fragB(const __bf16* p, int ld) {
  int lane = threadIdx.x & 31;
  const __bf16* row = p + (size_t)(lane & 15) * ld + ((lane >> 4) ? 8 : 0);
  return cat16(*(const v8bf*)(row), *(const v8bf*)(row + 16));
}

// ---------------------------------------------------------------------------
// fp32 -> bf16 streaming convert
// ---------------------------------------------------------------------------
__global__ void cvt_f32_to_bf16(const float* __restrict__ in,
                                __bf16* __restrict__ out, int n) {
  int stride = gridDim.x * blockDim.x * 4;
  for (int i = (blockIdx.x * blockDim.x + threadIdx.x) * 4; i < n; i += stride) {
    float4 f = *(const float4*)(in + i);
    v4bf o;
    o[0] = (__bf16)f.x; o[1] = (__bf16)f.y; o[2] = (__bf16)f.z; o[3] = (__bf16)f.w;
    *(v4bf*)(out + i) = o;
  }
}

// ---------------------------------------------------------------------------
// C = A(MxK bf16) @ W^T(W is NxK bf16) + bias, WMMA tiled.
// MODE 0: bf16 out scattered to [B,H,S,DK]        (Q/K projections)
// MODE 1: fp32 out row-major MxN                  (output projection)
// MODE 2: bf16 out scattered to [B,H,DK,SEQ]      (V projection, transposed,
//                                                  packed b128 stores)
// Block = 128 threads = 4 waves (2x2); wave tile 64x32 -> block tile 128x64.
// ---------------------------------------------------------------------------
template <int MODE>
__global__ __launch_bounds__(128)
void gemm_bias_kernel(const __bf16* __restrict__ A, const __bf16* __restrict__ W,
                      const float* __restrict__ bias, void* __restrict__ outp,
                      int M, int N, int K) {
  int wave = threadIdx.x >> 5;
  int lane = threadIdx.x & 31;
  int m0 = blockIdx.y * 128 + (wave >> 1) * 64;
  int n0 = blockIdx.x * 64 + (wave & 1) * 32;

  v8f c[4][2] = {};
  for (int k0 = 0; k0 < K; k0 += 32) {
    v16bf a0 = load_fragA(A + (size_t)(m0 +  0) * K + k0, K);
    v16bf a1 = load_fragA(A + (size_t)(m0 + 16) * K + k0, K);
    v16bf a2 = load_fragA(A + (size_t)(m0 + 32) * K + k0, K);
    v16bf a3 = load_fragA(A + (size_t)(m0 + 48) * K + k0, K);
    v16bf b0 = load_fragB(W + (size_t)n0 * K + k0, K);
    v16bf b1 = load_fragB(W + (size_t)(n0 + 16) * K + k0, K);
    c[0][0] = wmma_bf16(a0, b0, c[0][0]);
    c[0][1] = wmma_bf16(a0, b1, c[0][1]);
    c[1][0] = wmma_bf16(a1, b0, c[1][0]);
    c[1][1] = wmma_bf16(a1, b1, c[1][1]);
    c[2][0] = wmma_bf16(a2, b0, c[2][0]);
    c[2][1] = wmma_bf16(a2, b1, c[2][1]);
    c[3][0] = wmma_bf16(a3, b0, c[3][0]);
    c[3][1] = wmma_bf16(a3, b1, c[3][1]);
  }

  int lane15 = lane & 15;
  int mb = (lane >> 4) ? 8 : 0;
#pragma unroll
  for (int i = 0; i < 4; ++i)
#pragma unroll
    for (int j = 0; j < 2; ++j) {
      int nj = n0 + j * 16 + lane15;
      float bv = bias[nj];
      int mi0 = m0 + i * 16 + mb;
      if constexpr (MODE == 2) {
        // V^T: 8 consecutive s-positions pack into one 16B store
        int b = mi0 >> 11, s0 = mi0 & (SEQ - 1);
        int h = nj >> 6, d = nj & (DK - 1);
        v8bf pk;
#pragma unroll
        for (int r = 0; r < 8; ++r) pk[r] = (__bf16)(c[i][j][r] + bv);
        *(v8bf*)((__bf16*)outp + ((size_t)(b * NH + h) * DK + d) * SEQ + s0) = pk;
      } else {
#pragma unroll
        for (int r = 0; r < 8; ++r) {
          int mi = mi0 + r;
          float val = c[i][j][r] + bv;
          if constexpr (MODE == 0) {
            int b = mi >> 11, s = mi & (SEQ - 1);
            int h = nj >> 6, d = nj & (DK - 1);
            ((__bf16*)outp)[(((size_t)(b * NH + h)) * SEQ + s) * DK + d] = (__bf16)val;
          } else {
            ((float*)outp)[(size_t)mi * N + nj] = val;
          }
        }
      }
    }
}

// ---------------------------------------------------------------------------
// Flash attention, causal, LDS-free. One wave owns a 32-row query tile (two
// 16-query sub-tiles sharing every K/V fragment) and streams 32-key tiles.
// Computes S^T = K·Q^T so the probability C-fragment is bitwise the
// A-fragment layout needed for P·V (register-only conversion). Q is
// pre-scaled by log2(e)/sqrt(dk) so softmax uses raw v_exp (base-2).
// Causal masking runs only in the single diagonal tile.
// ---------------------------------------------------------------------------
__global__ __launch_bounds__(128)
void flash_attn_kernel(const __bf16* __restrict__ Qp, const __bf16* __restrict__ Kp,
                       const __bf16* __restrict__ Vt, __bf16* __restrict__ Xa) {
  int wave = threadIdx.x >> 5;
  int lane = threadIdx.x & 31;
  int lane15 = lane & 15;
  int mb = (lane >> 4) ? 8 : 0;

  int tile = blockIdx.x * 4 + wave;   // 0 .. B*NH*(SEQ/32)-1
  int qt = tile & 63;                 // SEQ/32 = 64
  int bh = tile >> 6;
  int h = bh & (NH - 1);
  int b = bh >> 4;

  const __bf16* Qb = Qp + ((size_t)(b * NH + h)) * SEQ * DK;  // [S, DK]
  const __bf16* Kb = Kp + ((size_t)(b * NH + h)) * SEQ * DK;  // [S, DK]
  const __bf16* Vb = Vt + ((size_t)(b * NH + h)) * DK * SEQ;  // [DK, S]
  int q0 = qt * 32;

  // Q^T as B-fragments (n = query, k = d), loaded once and pre-scaled by
  // log2(e)/sqrt(DK) so scores are already in the base-2 domain.
  const float qscale = 0.125f * 1.44269504088896340736f;
  v16bf qf[2][2];
#pragma unroll
  for (int qs = 0; qs < 2; ++qs)
#pragma unroll
    for (int dc = 0; dc < 2; ++dc) {
      v16bf t = load_fragB(Qb + (size_t)(q0 + 16 * qs) * DK + 32 * dc, DK);
#pragma unroll
      for (int e = 0; e < 16; ++e) t[e] = (__bf16)((float)t[e] * qscale);
      qf[qs][dc] = t;
    }

  v8f o[2][4] = {};                   // [qsub][d-subtile]
  float mrun[2] = {-1e30f, -1e30f};
  float lrun[2] = {0.0f, 0.0f};
  v16bf pa[2];

  auto step = [&](auto MASKED_, int n0) {
    constexpr bool MASKED = decltype(MASKED_)::value;
    // ---- S^T = K(keys x d) · Q^T(d x queries): 8 WMMAs for 32x32 scores
    v16bf ka0 = load_fragA(Kb + (size_t)n0 * DK, DK);
    v16bf ka1 = load_fragA(Kb + (size_t)n0 * DK + 32, DK);
    v16bf ka2 = load_fragA(Kb + (size_t)(n0 + 16) * DK, DK);
    v16bf ka3 = load_fragA(Kb + (size_t)(n0 + 16) * DK + 32, DK);
    v8f st[2][2];                     // [qsub][ksub]
#pragma unroll
    for (int qs = 0; qs < 2; ++qs) {
      v8f z = {};
      st[qs][0] = wmma_bf16(ka0, qf[qs][0], z);
      st[qs][0] = wmma_bf16(ka1, qf[qs][1], st[qs][0]);
      st[qs][1] = wmma_bf16(ka2, qf[qs][0], z);
      st[qs][1] = wmma_bf16(ka3, qf[qs][1], st[qs][1]);
    }

    // ---- online softmax per query sub-tile (lane owns one query column)
#pragma unroll
    for (int qs = 0; qs < 2; ++qs) {
      int qg = q0 + 16 * qs + lane15;
      float x0[8], x1[8];
      float tmax = -1e30f;
#pragma unroll
      for (int r = 0; r < 8; ++r) {
        float a0 = st[qs][0][r];
        float a1 = st[qs][1][r];
        if constexpr (MASKED) {
          if (n0 + mb + r      > qg) a0 = -1e9f;
          if (n0 + 16 + mb + r > qg) a1 = -1e9f;
        }
        x0[r] = a0; x1[r] = a1;
        tmax = fmaxf(tmax, fmaxf(a0, a1));
      }
      tmax = fmaxf(tmax, __shfl_xor(tmax, 16, 32));  // other half of key range

      float mnew = fmaxf(mrun[qs], tmax);
      float corr = exp2f(mrun[qs] - mnew);
      mrun[qs] = mnew;

      float tsum = 0.0f;
      v16bf p;
#pragma unroll
      for (int r = 0; r < 8; ++r) {
        float e0 = exp2f(x0[r] - mnew);
        float e1 = exp2f(x1[r] - mnew);
        tsum += e0 + e1;
        p[r]     = (__bf16)e0;        // keys kb..kb+7     (kb == mb)
        p[r + 8] = (__bf16)e1;        // keys 16+kb..16+kb+7
      }
      pa[qs] = p;
      tsum += __shfl_xor(tsum, 16, 32);
      lrun[qs] = lrun[qs] * corr + tsum;

      // rescale O rows: broadcast corr(query mb+r) from lane mb+r
#pragma unroll
      for (int r = 0; r < 8; ++r) {
        float cr = __shfl(corr, mb + r, 32);
        o[qs][0][r] *= cr; o[qs][1][r] *= cr;
        o[qs][2][r] *= cr; o[qs][3][r] *= cr;
      }
    }

    // ---- O += P · V: V fragments shared by both query sub-tiles
#pragma unroll
    for (int f = 0; f < 4; ++f) {
      v16bf vf = load_fragB(Vb + (size_t)(f * 16) * SEQ + n0, SEQ);
      o[0][f] = wmma_bf16(pa[0], vf, o[0][f]);
      o[1][f] = wmma_bf16(pa[1], vf, o[1][f]);
    }
  };

  for (int n0 = 0; n0 < q0; n0 += 32) step(std::false_type{}, n0);
  step(std::true_type{}, q0);        // single diagonal (masked) tile

  // ---- normalize and store to [B, S, H*DK] bf16 (row-major for final GEMM)
#pragma unroll
  for (int qs = 0; qs < 2; ++qs) {
    float invq = 1.0f / lrun[qs];
#pragma unroll
    for (int r = 0; r < 8; ++r) {
      float ir = __shfl(invq, mb + r, 32);
      int srow = q0 + 16 * qs + mb + r;
#pragma unroll
      for (int f = 0; f < 4; ++f)
        Xa[(((size_t)(b * SEQ) + srow) * NH + h) * DK + f * 16 + lane15] =
            (__bf16)(o[qs][f][r] * ir);
    }
  }
}

// ---------------------------------------------------------------------------
extern "C" void kernel_launch(void* const* d_in, const int* in_sizes, int n_in,
                              void* d_out, int out_size, void* d_ws, size_t ws_size,
                              hipStream_t stream) {
  (void)in_sizes; (void)n_in; (void)out_size; (void)ws_size;
  const float* q    = (const float*)d_in[0];
  const float* k    = (const float*)d_in[1];
  const float* v    = (const float*)d_in[2];
  // d_in[3] = mask: causal tril, handled analytically in the kernel
  const float* wq_w = (const float*)d_in[4];
  const float* wq_b = (const float*)d_in[5];
  const float* wk_w = (const float*)d_in[6];
  const float* wk_b = (const float*)d_in[7];
  const float* wv_w = (const float*)d_in[8];
  const float* wv_b = (const float*)d_in[9];
  const float* wo_w = (const float*)d_in[10];
  const float* wo_b = (const float*)d_in[11];

  const size_t NE = (size_t)BATCH * SEQ * DM;  // 4,194,304 activation elems
  const size_t NW = (size_t)DM * DM;           // 1,048,576 weight elems

  char* ws = (char*)d_ws;
  size_t off = 0;
  auto carve = [&](size_t bytes) -> void* {
    void* p = ws + off;
    off += (bytes + 255) & ~(size_t)255;
    return p;
  };
  __bf16* qb  = (__bf16*)carve(NE * 2);
  __bf16* kb  = (__bf16*)carve(NE * 2);
  __bf16* vb  = (__bf16*)carve(NE * 2);
  __bf16* wqb = (__bf16*)carve(NW * 2);
  __bf16* wkb = (__bf16*)carve(NW * 2);
  __bf16* wvb = (__bf16*)carve(NW * 2);
  __bf16* wob = (__bf16*)carve(NW * 2);
  __bf16* Qp  = (__bf16*)carve(NE * 2);   // [B,H,S,DK]
  __bf16* Kp  = (__bf16*)carve(NE * 2);   // [B,H,S,DK]
  __bf16* Vp  = (__bf16*)carve(NE * 2);   // [B,H,DK,SEQ]  (transposed)
  __bf16* Xa  = (__bf16*)carve(NE * 2);   // [B,S,H*DK]

  auto cvt = [&](const float* in, __bf16* out, size_t n) {
    int threads = 256;
    int blocks = (int)((n / 4 + threads - 1) / threads);
    if (blocks > 2048) blocks = 2048;
    cvt_f32_to_bf16<<<blocks, threads, 0, stream>>>(in, out, (int)n);
  };
  cvt(q, qb, NE);
  cvt(k, kb, NE);
  cvt(v, vb, NE);
  cvt(wq_w, wqb, NW);
  cvt(wk_w, wkb, NW);
  cvt(wv_w, wvb, NW);
  cvt(wo_w, wob, NW);

  dim3 gg(DM / 64, (BATCH * SEQ) / 128);  // (16, 32), block tile 128x64
  gemm_bias_kernel<0><<<gg, 128, 0, stream>>>(qb, wqb, wq_b, Qp, BATCH * SEQ, DM, DM);
  gemm_bias_kernel<0><<<gg, 128, 0, stream>>>(kb, wkb, wk_b, Kp, BATCH * SEQ, DM, DM);
  gemm_bias_kernel<2><<<gg, 128, 0, stream>>>(vb, wvb, wv_b, Vp, BATCH * SEQ, DM, DM);

  int atiles = BATCH * NH * (SEQ / 32);   // 2048 query tiles, 4 waves/block
  flash_attn_kernel<<<atiles / 4, 128, 0, stream>>>(Qp, Kp, Vp, Xa);

  gemm_bias_kernel<1><<<gg, 128, 0, stream>>>(Xa, wob, wo_b, d_out, BATCH * SEQ, DM, DM);
}
